// ResBlock_89696097010023
// MI455X (gfx1250) — matrix operand order repaired and checked by reading, not compiled, not run on hardware
//
#include <hip/hip_runtime.h>
#include <hip/hip_bf16.h>
#include <math.h>

// ---------------------------------------------------------------------------
// ResBlock: op2 = conv3x3(x, w2); a = act(op2) (PReLU / ReLU / tanh by c%3,
// channel permutation folded into w3); out = conv3x3(a, w3)*0.1 + x
// Implicit-GEMM with V_WMMA_F32_16X16X32_BF16 (wave32). Both operands staged
// in LDS per ci-chunk via GLOBAL_LOAD_ASYNC_TO_LDS_B128 (ASYNCcnt-tracked);
// the 9-tap inner loop is pure ds_load + WMMA.
// ---------------------------------------------------------------------------

typedef __attribute__((ext_vector_type(16))) __bf16 v16bf;
typedef __attribute__((ext_vector_type(8)))  __bf16 v8bf;
typedef __attribute__((ext_vector_type(8)))  float  v8f;

#define C_CH 256
#define B_N  16
#define H_SP 96
#define W_SP 96
#define HP   98
#define WP   98
#define KCHUNKS 72            // 9 taps * (256/32) ci-chunks
#define NCIC 8                // ci-chunks per tap
#define RES_SCALE 0.1f

static constexpr long long PAD_ELEMS = (long long)B_N * HP * WP * C_CH;   // 39,337,984
static constexpr long long PAD_BYTES = PAD_ELEMS * 2;                     // 78,675,968
static constexpr int       WF_ELEMS  = 16 * KCHUNKS * 32 * 16;            // 589,824
static constexpr long long WF_BYTES  = (long long)WF_ELEMS * 2;

// ---------------------------------------------------------------------------
// Async 16-byte global -> LDS copy (CDNA5 TDM-free async path, ASYNCcnt).
// ldsOff: wave-relative LDS byte offset (low 32 bits of generic pointer).
// ---------------------------------------------------------------------------
__device__ __forceinline__ void async_copy_b128(unsigned ldsOff, const void* g) {
    asm volatile("global_load_async_to_lds_b128 %0, %1, off"
                 :: "v"(ldsOff), "v"(g) : "memory");
}
__device__ __forceinline__ void wait_async0() {
    asm volatile("s_wait_asynccnt 0" ::: "memory");
}

// ---------------------------------------------------------------------------
// Prep: x (f32 NCHW) -> xb (bf16, padded, channel-last), zero borders.
// ---------------------------------------------------------------------------
__global__ void pack_x_kernel(const float* __restrict__ x, __bf16* __restrict__ xb) {
    long long idx = (long long)blockIdx.x * blockDim.x + threadIdx.x;
    if (idx >= PAD_ELEMS) return;
    int c  = (int)(idx & (C_CH - 1));
    long long t = idx >> 8;
    int wp = (int)(t % WP); t /= WP;
    int hp = (int)(t % HP); t /= HP;
    int b  = (int)t;
    float v = 0.0f;
    if (hp >= 1 && hp <= H_SP && wp >= 1 && wp <= W_SP) {
        v = x[(((long long)b * C_CH + c) * H_SP + (hp - 1)) * W_SP + (wp - 1)];
    }
    xb[idx] = (__bf16)v;
}

// ---------------------------------------------------------------------------
// Prep: zero-fill a buffer (uint4 granularity)
// ---------------------------------------------------------------------------
__global__ void fill_zero_kernel(uint4* __restrict__ p, long long n16) {
    long long i = (long long)blockIdx.x * blockDim.x + threadIdx.x;
    if (i < n16) p[i] = make_uint4(0u, 0u, 0u, 0u);
}

// ---------------------------------------------------------------------------
// Prep: swizzle OIHW f32 weights into WMMA A-fragment-major bf16.
// A 16x32 lane layout used here (same contiguous k order as the B strip):
//   lanes 0-15  : V0-7 -> K 0..15
//   lanes 16-31 : V0-7 -> K 16..31
// Global K = tap*256 + ci, chunk = tap*8 + cic.
// permute!=0 folds the idx_p/idx_r/idx_t concat permutation into ci.
// ---------------------------------------------------------------------------
__global__ void pack_w_kernel(const float* __restrict__ w, __bf16* __restrict__ wf,
                              int permute) {
    int idx = blockIdx.x * blockDim.x + threadIdx.x;
    if (idx >= WF_ELEMS) return;
    int e    = idx & 1;
    int v    = (idx >> 1) & 7;
    int lane = (idx >> 4) & 31;
    int ch   = (idx >> 9) % KCHUNKS;
    int ct   = (idx >> 9) / KCHUNKS;       // co-tile (16 co each)

    int klocal = 2 * v + e + (lane < 16 ? 0 : 16);
    int tap = ch >> 3;                     // 0..8
    int ci  = (ch & 7) * 32 + klocal;      // natural channel of activated tensor
    int kh  = tap / 3, kw = tap % 3;
    int co  = ct * 16 + (lane & 15);

    int cim = ci;
    if (permute) {
        int g = ci % 3;
        cim = (g == 0 ? 0 : (g == 1 ? 86 : 171)) + ci / 3;   // original w3 ci index
    }
    wf[idx] = (__bf16)w[((co * C_CH + cim) * 3 + kh) * 3 + kw];
}

// ---------------------------------------------------------------------------
// Main implicit-GEMM conv kernel. 192 threads = 6 waves (2 co x 3 n).
// Workgroup tile: 128 co x 96 px. Wave: 64co x 32px = 8 acc tiles.
// Per ci-chunk stage, async-copy to LDS (no VGPR transit):
//   - B strip : 3 rows x 98 px x 32 ci                  (18.8 KB)
//   - A frags : 8 co-tiles x 9 taps x 32 lanes x 32 B   (72 KB)
// Inner 9-tap loop is pure LDS: 4 A + 2 B ds_load_b128 pairs (immediate
// offsets) feeding 8 WMMAs -- no global loads, no loadcnt waits.
// MODE 0: epilogue = PReLU/ReLU/tanh by co%3 -> bf16 padded channel-last dst.
// MODE 1: epilogue = acc*0.1 + x residual -> f32 NCHW.
// ---------------------------------------------------------------------------
template <int MODE>
__global__ __launch_bounds__(192)
void conv_gemm_kernel(const __bf16* __restrict__ src,    // padded channel-last bf16
                      const __bf16* __restrict__ wfrag,  // fragment-major weights
                      const float*  __restrict__ xres,   // MODE 1: residual (NCHW f32)
                      const float*  __restrict__ prelu,  // MODE 0: prelu weight
                      void*         __restrict__ dst) {
    __shared__ __bf16 strip[3 * WP * 32];                // 18,816 B
    __shared__ __bf16 aLds[8 * 9 * 32 * 16];             // 73,728 B

    const int tid    = threadIdx.x;
    const int lane   = tid & 31;
    const int wave   = tid >> 5;
    const int waveCo = wave & 1;          // 0..1 -> co offset 64*waveCo
    const int waveN  = wave >> 1;         // 0..2 -> px offset 32*waveN
    const int row    = blockIdx.x;        // 0 .. B*H-1
    const int b      = row / H_SP;
    const int h      = row % H_SP;
    const int coWG   = blockIdx.y * 128;
    const int n      = lane & 15;         // pixel within 16-px subtile
    const int half   = lane >> 4;         // K-half (B) / M-half (C/D)

    v8f acc[4][2] = {};

    const int tileIdx0 = coWG >> 4;                      // first of 8 WG co-tiles
    const long long rowBase = ((long long)b * HP + h) * WP;
    const int wBase = waveN * 32 + n;                    // pixel (nt=0)

    const v16bf* __restrict__ wfg = (const v16bf*)wfrag; // fragment granularity

    // Wave-relative LDS byte offsets of the staging buffers
    const unsigned stripOff = (unsigned)(unsigned long long)(const void*)&strip[0];
    const unsigned aLdsOff  = (unsigned)(unsigned long long)(const void*)&aLds[0];

    // LDS base pointers (per-tap offsets are compile-time constants)
    const __bf16* bP  = &strip[wBase * 32 + half * 16];
    const int lt0 = waveCo * 4;                          // wave's first local tile
    const __bf16* aB0 = &aLds[((lt0 + 0) * 9 * 32 + lane) * 16];
    const __bf16* aB1 = &aLds[((lt0 + 1) * 9 * 32 + lane) * 16];
    const __bf16* aB2 = &aLds[((lt0 + 2) * 9 * 32 + lane) * 16];
    const __bf16* aB3 = &aLds[((lt0 + 3) * 9 * 32 + lane) * 16];

    for (int cic = 0; cic < NCIC; ++cic) {
        // ---- stage B strip: 3 rows x 98 px x 32 ci (64 B per pixel) --------
        for (int u = tid; u < 3 * WP; u += 192) {
            const int r  = u / WP;
            const int px = u % WP;
            const char* g = (const char*)
                (src + (rowBase + (long long)r * WP + px) * C_CH + cic * 32);
            const unsigned lo = stripOff + (unsigned)(u * 64);
            async_copy_b128(lo,      g);
            async_copy_b128(lo + 16, g + 16);
            async_copy_b128(lo + 32, g + 32);
            async_copy_b128(lo + 48, g + 48);
            if (cic + 1 < NCIC) __builtin_prefetch(g + 64, 0, 0);
        }
        // ---- stage A frags: 8 tiles x 9 taps x 32 lanes (exactly 12/thread)
        for (int u = tid; u < 8 * 9 * 32; u += 192) {
            const int l   = u & 31;
            const int tap = (u >> 5) % 9;
            const int til = (u >> 5) / 9;
            const char* gsrc = (const char*)
                &wfg[((tileIdx0 + til) * KCHUNKS + tap * NCIC + cic) * 32 + l];
            const unsigned lo = aLdsOff + (unsigned)(u * 32);
            async_copy_b128(lo,      gsrc);
            async_copy_b128(lo + 16, gsrc + 16);
            if (cic + 1 < NCIC) __builtin_prefetch(gsrc + 1024, 0, 0);
        }
        wait_async0();                     // this wave's async copies landed
        __syncthreads();                   // all waves' copies visible

        // ---- 9 taps, everything from LDS -----------------------------------
        #pragma unroll
        for (int tap = 0; tap < 9; ++tap) {
            const int kh = tap / 3;
            const int kw = tap % 3;

            v16bf a0 = *(const v16bf*)(aB0 + tap * 512);
            v16bf a1 = *(const v16bf*)(aB1 + tap * 512);
            v16bf a2 = *(const v16bf*)(aB2 + tap * 512);
            v16bf a3 = *(const v16bf*)(aB3 + tap * 512);

            const __bf16* bp = bP + (kh * WP + kw) * 32;
            v16bf b0 = *(const v16bf*)bp;
            v16bf b1 = *(const v16bf*)(bp + 16 * 32);     // +16 pixels

            acc[0][0] = __builtin_amdgcn_wmma_f32_16x16x32_bf16(false, a0, false, b0, (short)0, acc[0][0], false, false);
            acc[0][1] = __builtin_amdgcn_wmma_f32_16x16x32_bf16(false, a0, false, b1, (short)0, acc[0][1], false, false);
            acc[1][0] = __builtin_amdgcn_wmma_f32_16x16x32_bf16(false, a1, false, b0, (short)0, acc[1][0], false, false);
            acc[1][1] = __builtin_amdgcn_wmma_f32_16x16x32_bf16(false, a1, false, b1, (short)0, acc[1][1], false, false);
            acc[2][0] = __builtin_amdgcn_wmma_f32_16x16x32_bf16(false, a2, false, b0, (short)0, acc[2][0], false, false);
            acc[2][1] = __builtin_amdgcn_wmma_f32_16x16x32_bf16(false, a2, false, b1, (short)0, acc[2][1], false, false);
            acc[3][0] = __builtin_amdgcn_wmma_f32_16x16x32_bf16(false, a3, false, b0, (short)0, acc[3][0], false, false);
            acc[3][1] = __builtin_amdgcn_wmma_f32_16x16x32_bf16(false, a3, false, b1, (short)0, acc[3][1], false, false);
        }
        __syncthreads();
    }

    // ---- epilogue -----------------------------------------------------------
    if (MODE == 0) {
        const float pw = prelu[0];
        __bf16* adst = (__bf16*)dst;
        #pragma unroll
        for (int ct = 0; ct < 4; ++ct) {
            #pragma unroll
            for (int nt = 0; nt < 2; ++nt) {
                const int co0 = coWG + waveCo * 64 + ct * 16 + half * 8;  // 8 consecutive co
                const int wpx = waveN * 32 + nt * 16 + n;
                v8bf ov;
                #pragma unroll
                for (int r = 0; r < 8; ++r) {
                    const int co = co0 + r;
                    float vv = acc[ct][nt][r];
                    const int g = co % 3;
                    if (g == 0)      vv = (vv >= 0.0f) ? vv : pw * vv;
                    else if (g == 1) vv = fmaxf(vv, 0.0f);
                    else             vv = tanhf(vv);
                    ov[r] = (__bf16)vv;
                }
                __bf16* p = adst + ((((long long)b * HP + (h + 1)) * WP + (wpx + 1)) * C_CH + co0);
                *(v8bf*)p = ov;
            }
        }
    } else {
        float* outp = (float*)dst;
        #pragma unroll
        for (int ct = 0; ct < 4; ++ct) {
            #pragma unroll
            for (int nt = 0; nt < 2; ++nt) {
                const int co0 = coWG + waveCo * 64 + ct * 16 + half * 8;
                const int wpx = waveN * 32 + nt * 16 + n;
                #pragma unroll
                for (int r = 0; r < 8; ++r) {
                    const int co = co0 + r;
                    const long long o =
                        (((long long)b * C_CH + co) * H_SP + h) * W_SP + wpx;
                    outp[o] = acc[ct][nt][r] * RES_SCALE + xres[o];
                }
            }
        }
    }
}

// ---------------------------------------------------------------------------
// Host launcher
// ---------------------------------------------------------------------------
extern "C" void kernel_launch(void* const* d_in, const int* in_sizes, int n_in,
                              void* d_out, int out_size, void* d_ws, size_t ws_size,
                              hipStream_t stream) {
    (void)in_sizes; (void)n_in; (void)out_size; (void)ws_size;

    const float* x     = (const float*)d_in[0];
    const float* w2    = (const float*)d_in[1];
    const float* w3    = (const float*)d_in[2];
    const float* prelu = (const float*)d_in[3];
    float*       out   = (float*)d_out;

    char*   ws   = (char*)d_ws;
    __bf16* xb   = (__bf16*)ws;
    __bf16* apad = (__bf16*)(ws + PAD_BYTES);
    __bf16* w2f  = (__bf16*)(ws + 2 * PAD_BYTES);
    __bf16* w3f  = (__bf16*)(ws + 2 * PAD_BYTES + WF_BYTES);

    // Prep: pack x to padded channel-last bf16
    pack_x_kernel<<<dim3((unsigned)((PAD_ELEMS + 255) / 256)), dim3(256), 0, stream>>>(x, xb);
    // Prep: zero a_pad (borders must be 0; interior overwritten by conv2 epilogue)
    {
        long long n16 = PAD_BYTES / 16;
        fill_zero_kernel<<<dim3((unsigned)((n16 + 255) / 256)), dim3(256), 0, stream>>>((uint4*)apad, n16);
    }
    // Prep: swizzle weights into WMMA fragment layout
    pack_w_kernel<<<dim3((WF_ELEMS + 255) / 256), dim3(256), 0, stream>>>(w2, w2f, 0);
    pack_w_kernel<<<dim3((WF_ELEMS + 255) / 256), dim3(256), 0, stream>>>(w3, w3f, 1);

    // conv2 + fused activation -> a_pad (bf16, padded, channel-last)
    dim3 grid(B_N * H_SP, C_CH / 128), block(192);
    conv_gemm_kernel<0><<<grid, block, 0, stream>>>(xb, w2f, nullptr, prelu, (void*)apad);
    // conv3 + fused residual -> out (f32, NCHW)
    conv_gemm_kernel<1><<<grid, block, 0, stream>>>(apad, w3f, x, nullptr, (void*)out);
}
